// LabelingLayer_21105469293054
// MI455X (gfx1250) — matrix operand order repaired and checked by reading, not compiled, not run on hardware
//
#include <hip/hip_runtime.h>

// DEC / t-SNE soft-assignment:  q = rownorm( 1 / (1 + ||x_n - c_k||^2) )
// cross term via v_wmma_f32_16x16x32_bf16 (fp32 accumulate), norms in fp32.
// Pipelined: LDS double-buffer for cluster chunks (1 barrier/chunk, global
// loads overlap WMMA) + register double-buffer for B fragments (ds latency
// hides under the XDL pipe).

typedef __attribute__((ext_vector_type(16))) __bf16 v16bf;
typedef __attribute__((ext_vector_type(8)))  __bf16 v8bf;
typedef __attribute__((ext_vector_type(8)))  float  v8f;

#define D_DIM          256
#define K_DIM          256
#define ROWS_PER_BLOCK 128
#define THREADS        256
#define KCHUNK         32
#define NCHUNKS        (D_DIM / KCHUNK)
#define LDSB_STRIDE    40   // 32 bf16 data + 8 bf16 pad -> 20-dword row shift, conflict-free

__global__ __launch_bounds__(THREADS)
void dec_soft_assign_kernel(const float* __restrict__ x,
                            const float* __restrict__ clusters,
                            float* __restrict__ out)
{
    __shared__ __bf16 ldsB[2][K_DIM * LDSB_STRIDE]; // 2 x 20 KB bf16 chunks [256 cols][32 k]
    __shared__ float  ldsCsq[K_DIM];                // 1 KB: ||c_k||^2

    const int tid  = threadIdx.x;
    const int lane = tid & 31;
    const int wv   = tid >> 5;            // wave 0..7 -> 16-row strip
    const int m    = lane & 15;           // row within strip / col within B tile
    const int hlf  = lane >> 4;           // lane half selects k sub-group
    const int koff = hlf * 8;             // A layout: {0..7,16..23} vs {8..15,24..31}
    const int rowBase = blockIdx.x * ROWS_PER_BLOCK;

    v8f acc[16];
    #pragma unroll
    for (int nt = 0; nt < 16; ++nt)
        acc[nt] = (v8f){0.f,0.f,0.f,0.f,0.f,0.f,0.f,0.f};

    float xsq = 0.0f;   // partial ||x_row||^2 (this lane's k-subset)
    float csq = 0.0f;   // ||clusters[tid]||^2 (full, accumulated over chunks)

    const float* __restrict__ arow =
        x + (size_t)(rowBase + wv * 16 + m) * D_DIM + koff;
    const float* __restrict__ crow = clusters + (size_t)tid * D_DIM;

    // cooperative fp32->bf16 convert of clusters[:, kb*32 : kb*32+32] into LDS buf
    auto convert_chunk = [&](int kb, int buf) {
        const float* cp = crow + kb * KCHUNK;
        __bf16* lp = &ldsB[buf][tid * LDSB_STRIDE];
        #pragma unroll
        for (int c = 0; c < 4; ++c) {
            float4 f0 = *(const float4*)(cp + c * 8);
            float4 f1 = *(const float4*)(cp + c * 8 + 4);
            csq += f0.x*f0.x + f0.y*f0.y + f0.z*f0.z + f0.w*f0.w
                 + f1.x*f1.x + f1.y*f1.y + f1.z*f1.z + f1.w*f1.w;
            v8bf h;
            h[0] = (__bf16)f0.x; h[1] = (__bf16)f0.y;
            h[2] = (__bf16)f0.z; h[3] = (__bf16)f0.w;
            h[4] = (__bf16)f1.x; h[5] = (__bf16)f1.y;
            h[6] = (__bf16)f1.z; h[7] = (__bf16)f1.w;
            *(v8bf*)(lp + c * 8) = h;
        }
    };

    convert_chunk(0, 0);
    __syncthreads();

    #pragma unroll 1
    for (int kb = 0; kb < NCHUNKS; ++kb) {
        const int buf = kb & 1;

        // stage next chunk into the other LDS buffer (overlaps with WMMA below)
        if (kb < NCHUNKS - 1)
            convert_chunk(kb + 1, buf ^ 1);

        // -- A fragment: 16 rows x 32 k, ISA 16-bit A 16x32 lane layout --
        const int d0 = kb * KCHUNK;
        float4 f0 = *(const float4*)(arow + d0);        // k = koff+0..3
        float4 f1 = *(const float4*)(arow + d0 + 4);    // k = koff+4..7
        float4 f2 = *(const float4*)(arow + d0 + 16);   // k = koff+16..19
        float4 f3 = *(const float4*)(arow + d0 + 20);   // k = koff+20..23
        xsq += f0.x*f0.x + f0.y*f0.y + f0.z*f0.z + f0.w*f0.w
             + f1.x*f1.x + f1.y*f1.y + f1.z*f1.z + f1.w*f1.w
             + f2.x*f2.x + f2.y*f2.y + f2.z*f2.z + f2.w*f2.w
             + f3.x*f3.x + f3.y*f3.y + f3.z*f3.z + f3.w*f3.w;
        v16bf a;
        a[0]  = (__bf16)f0.x; a[1]  = (__bf16)f0.y;
        a[2]  = (__bf16)f0.z; a[3]  = (__bf16)f0.w;
        a[4]  = (__bf16)f1.x; a[5]  = (__bf16)f1.y;
        a[6]  = (__bf16)f1.z; a[7]  = (__bf16)f1.w;
        a[8]  = (__bf16)f2.x; a[9]  = (__bf16)f2.y;
        a[10] = (__bf16)f2.z; a[11] = (__bf16)f2.w;
        a[12] = (__bf16)f3.x; a[13] = (__bf16)f3.y;
        a[14] = (__bf16)f3.z; a[15] = (__bf16)f3.w;

        // -- 16 WMMA tiles over 256 cluster columns, B regs double-buffered --
        // B layout (32x16 bf16): lanes 0-15 col n hold K={0..7,16..23}, lanes 16-31 K={8..15,24..31}
        const __bf16* bbase = &ldsB[buf][(size_t)m * LDSB_STRIDE + hlf * 8];
        v8bf blo[2], bhi[2];
        blo[0] = *(const v8bf*)(bbase);
        bhi[0] = *(const v8bf*)(bbase + 16);
        #pragma unroll
        for (int nt = 0; nt < 16; ++nt) {
            const int cur = nt & 1;
            if (nt < 15) {
                const __bf16* bp = bbase + (nt + 1) * 16 * LDSB_STRIDE;
                blo[cur ^ 1] = *(const v8bf*)(bp);
                bhi[cur ^ 1] = *(const v8bf*)(bp + 16);
            }
            v16bf b;
            #pragma unroll
            for (int j = 0; j < 8; ++j) {
                b[j]     = blo[cur][j];
                b[8 + j] = bhi[cur][j];
            }
            acc[nt] = __builtin_amdgcn_wmma_f32_16x16x32_bf16(
                false, a, false, b, (short)0, acc[nt], false, false);
        }
        __syncthreads();   // next iter reads buf^1 / overwrites buf
    }

    ldsCsq[tid] = csq;
    __syncthreads();

    // full ||x||^2 per row: lane pair (L, L^16) covers complementary k-subsets
    const float xsq_full = xsq + __shfl_xor(xsq, 16, 32);

    float cs[16];
    #pragma unroll
    for (int nt = 0; nt < 16; ++nt) cs[nt] = ldsCsq[nt * 16 + m];

    // C-tile layout: VGPR v -> row v (lanes 0-15) / row v+8 (lanes 16-31), col = nt*16 + m
    #pragma unroll
    for (int v = 0; v < 8; ++v) {
        const int rowLocal = v + hlf * 8;
        const float xs = __shfl(xsq_full, rowLocal, 32);  // lanes 0..15 hold x_sq[row]
        float q[16];
        float partial = 0.0f;
        #pragma unroll
        for (int nt = 0; nt < 16; ++nt) {
            float d2 = xs - 2.0f * acc[nt][v] + cs[nt];
            d2 = fmaxf(d2, 0.0f);                 // guard tiny negatives from rounding
            const float qv = 1.0f / (1.0f + d2);  // ALPHA=1: power (a+1)/2 == 1
            q[nt] = qv;
            partial += qv;
        }
        // row-sum across the 16 lanes of this half (xor 16 would mix rows)
        #pragma unroll
        for (int off = 1; off < 16; off <<= 1)
            partial += __shfl_xor(partial, off, 32);
        const float inv = 1.0f / partial;

        float* orow = out + (size_t)(rowBase + wv * 16 + rowLocal) * K_DIM;
        #pragma unroll
        for (int nt = 0; nt < 16; ++nt)
            orow[nt * 16 + m] = q[nt] * inv;      // each half-wave: 64B contiguous
    }
}

extern "C" void kernel_launch(void* const* d_in, const int* in_sizes, int n_in,
                              void* d_out, int out_size, void* d_ws, size_t ws_size,
                              hipStream_t stream) {
    (void)n_in; (void)out_size; (void)d_ws; (void)ws_size;
    const float* x        = (const float*)d_in[0];
    const float* clusters = (const float*)d_in[1];
    float* out            = (float*)d_out;
    const int N = in_sizes[0] / D_DIM;              // 262144
    const int nblocks = N / ROWS_PER_BLOCK;         // 2048
    dec_soft_assign_kernel<<<nblocks, THREADS, 0, stream>>>(x, clusters, out);
}
